// GraphAggregator_15187004358828
// MI455X (gfx1250) — matrix operand order repaired and checked by reading, not compiled, not run on hardware
//
#include <hip/hip_runtime.h>

typedef __attribute__((ext_vector_type(16))) _Float16 v16h;
typedef __attribute__((ext_vector_type(8)))  _Float16 v8h;
typedef __attribute__((ext_vector_type(8)))  float    v8f;
typedef __attribute__((ext_vector_type(4)))  float    v4f;

#define N_NODES_C   1000000
#define D_INP       128
#define H1_DIM      64
#define H2_DIM      256
#define GSD         128
#define N_GRAPHS_C  4096
#define N_TILES     (N_NODES_C / 16)   // 62500 exact
#define WPB         8                  // waves per block (wave32)
#define BLOCK_SZ    256

static __device__ __forceinline__ v8f wmma_f16(v16h a, v16h b, v8f c) {
    // v_wmma_f32_16x16x32_f16: 8 args (neg_a, A, neg_b, B, c_mod, C, reuse_a, reuse_b)
    return __builtin_amdgcn_wmma_f32_16x16x32_f16(false, a, false, b, (short)0, c, false, false);
}

// ---------------------------------------------------------------------------
// Kernel 0: zero the f32 graph-state accumulator in workspace
// ---------------------------------------------------------------------------
__global__ void zero_f32(float* __restrict__ p, int n) {
    int i = blockIdx.x * blockDim.x + threadIdx.x;
    if (i < n) p[i] = 0.0f;
}

// ---------------------------------------------------------------------------
// Kernel 1: MLP1 (X@W1 relu @W2) + sigmoid gating + atomic segment-sum
// One wave = one 16-node tile. Weights pre-swizzled into LDS B-fragments.
// ---------------------------------------------------------------------------
__global__ __launch_bounds__(BLOCK_SZ) void mlp1_gate_scatter(
    const float* __restrict__ X, const int* __restrict__ gidx,
    const float* __restrict__ W1, const float* __restrict__ b1,
    const float* __restrict__ W2, const float* __restrict__ b2,
    float* __restrict__ gacc)
{
    // B-fragment layout per v_wmma_f32_16x16x32_f16: frag = 512 halves,
    // addressed as [lane*16 + i]; lane&15 = N column, lane>>4 selects K half.
    __shared__ __align__(32) _Float16 sW1[16 * 512];          // (nt*4 + kt)
    __shared__ __align__(32) _Float16 sW2[32 * 512];          // (ct*2 + kt)
    __shared__ __align__(32) _Float16 sH1[WPB][16 * 64];      // per-wave staging
    __shared__ float sB1[H1_DIM];
    __shared__ float sB2[H2_DIM];

    const int tid = threadIdx.x;

    for (int idx = tid; idx < 16 * 512; idx += BLOCK_SZ) {
        int f = idx >> 9, l = (idx >> 4) & 31, i = idx & 15;
        int nt = f >> 2, kt = f & 3;
        int n = nt * 16 + (l & 15);
        int k = kt * 32 + ((i >> 3) << 4) + ((l >> 4) << 3) + (i & 7);
        sW1[idx] = (_Float16)W1[k * H1_DIM + n];
    }
    for (int idx = tid; idx < 32 * 512; idx += BLOCK_SZ) {
        int f = idx >> 9, l = (idx >> 4) & 31, i = idx & 15;
        int ct = f >> 1, kt = f & 1;
        int n = ct * 16 + (l & 15);
        int k = kt * 32 + ((i >> 3) << 4) + ((l >> 4) << 3) + (i & 7);
        sW2[idx] = (_Float16)W2[k * H2_DIM + n];
    }
    if (tid < H1_DIM) sB1[tid] = b1[tid];
    sB2[tid] = b2[tid];                       // BLOCK_SZ == H2_DIM == 256
    __syncthreads();

    const int wave  = tid >> 5;
    const int lane  = tid & 31;
    const int lrow  = lane & 15;
    const int khalf = lane >> 4;
    _Float16* myH1  = sH1[wave];

    const int totalWaves = gridDim.x * WPB;
    for (int t = blockIdx.x * WPB + wave; t < N_TILES; t += totalWaves) {
        const int m0 = t << 4;
        const float* xrow = X + (size_t)(m0 + lrow) * D_INP;

        // -------- A fragments for layer 1 (16x128 fp32 -> 4 x 16x32 f16) ----
        v16h a1[4];
        #pragma unroll
        for (int kt = 0; kt < 4; ++kt) {
            const float* p = xrow + kt * 32 + khalf * 8;
            v4f r0 = *(const v4f*)(p);        // K = khalf*8 + 0..3
            v4f r1 = *(const v4f*)(p + 4);    // K = khalf*8 + 4..7
            v4f r2 = *(const v4f*)(p + 16);   // K = 16 + khalf*8 + 0..3
            v4f r3 = *(const v4f*)(p + 20);   // K = 16 + khalf*8 + 4..7
            v16h a;
            #pragma unroll
            for (int j = 0; j < 4; ++j) {
                a[j]      = (_Float16)r0[j];
                a[4 + j]  = (_Float16)r1[j];
                a[8 + j]  = (_Float16)r2[j];
                a[12 + j] = (_Float16)r3[j];
            }
            a1[kt] = a;
        }

        // -------- layer 1: H1 = relu(X@W1 + b1), stage to LDS as f16 -------
        #pragma unroll
        for (int nt = 0; nt < 4; ++nt) {
            v8f acc = {};
            #pragma unroll
            for (int kt = 0; kt < 4; ++kt) {
                v16h b = *(const v16h*)&sW1[((nt << 2) | kt) * 512 + lane * 16];
                acc = wmma_f16(a1[kt], b, acc);
            }
            const float bias = sB1[nt * 16 + lrow];
            #pragma unroll
            for (int v = 0; v < 8; ++v) {
                float h = fmaxf(acc[v] + bias, 0.0f);
                myH1[(v + 8 * khalf) * 64 + nt * 16 + lrow] = (_Float16)h;
            }
        }

        // -------- re-fragment H1 (D-layout -> A-layout) via per-wave LDS ---
        v16h a2[2];
        #pragma unroll
        for (int kt = 0; kt < 2; ++kt) {
            const _Float16* p = myH1 + lrow * 64 + kt * 32 + khalf * 8;
            v8h lo = *(const v8h*)(p);
            v8h hi = *(const v8h*)(p + 16);
            v16h a;
            #pragma unroll
            for (int j = 0; j < 8; ++j) { a[j] = lo[j]; a[8 + j] = hi[j]; }
            a2[kt] = a;
        }

        int gbase[8];
        #pragma unroll
        for (int v = 0; v < 8; ++v)
            gbase[v] = gidx[m0 + v + 8 * khalf] * GSD;

        // -------- layer 2 in (gate, candidate) tile pairs + scatter --------
        #pragma unroll
        for (int nt = 0; nt < 8; ++nt) {
            v8f accG = {}, accC = {};
            #pragma unroll
            for (int kt = 0; kt < 2; ++kt) {
                v16h bg = *(const v16h*)&sW2[((nt << 1) | kt) * 512 + lane * 16];
                accG = wmma_f16(a2[kt], bg, accG);
                v16h bc = *(const v16h*)&sW2[(((nt + 8) << 1) | kt) * 512 + lane * 16];
                accC = wmma_f16(a2[kt], bc, accC);
            }
            const float bgv = sB2[nt * 16 + lrow];
            const float bcv = sB2[128 + nt * 16 + lrow];
            #pragma unroll
            for (int v = 0; v < 8; ++v) {
                float g = accG[v] + bgv;
                float c = accC[v] + bcv;
                float s = 1.0f / (1.0f + __expf(-g));
                unsafeAtomicAdd(&gacc[gbase[v] + nt * 16 + lrow], s * c);
            }
        }
    }
}

// ---------------------------------------------------------------------------
// Kernel 2: MLP2 readout: out = relu(gacc@W3 + b3)@W4 + b4   [4096,128]->[4096,16]
// ---------------------------------------------------------------------------
__global__ __launch_bounds__(BLOCK_SZ) void mlp2_readout(
    const float* __restrict__ gacc,
    const float* __restrict__ W3, const float* __restrict__ b3,
    const float* __restrict__ W4, const float* __restrict__ b4,
    float* __restrict__ out)
{
    __shared__ __align__(32) _Float16 sW3[8 * 512];       // (nt*4 + kt)
    __shared__ __align__(32) _Float16 sW4[512];
    __shared__ __align__(32) _Float16 sH3[WPB][16 * 32];
    __shared__ float sB3[32];
    __shared__ float sB4[16];

    const int tid = threadIdx.x;
    for (int idx = tid; idx < 8 * 512; idx += BLOCK_SZ) {
        int f = idx >> 9, l = (idx >> 4) & 31, i = idx & 15;
        int nt = f >> 2, kt = f & 3;
        int n = nt * 16 + (l & 15);
        int k = kt * 32 + ((i >> 3) << 4) + ((l >> 4) << 3) + (i & 7);
        sW3[idx] = (_Float16)W3[k * 32 + n];
    }
    for (int idx = tid; idx < 512; idx += BLOCK_SZ) {
        int l = (idx >> 4) & 31, i = idx & 15;
        int n = l & 15;
        int k = ((i >> 3) << 4) + ((l >> 4) << 3) + (i & 7);
        sW4[idx] = (_Float16)W4[k * 16 + n];
    }
    if (tid < 32) sB3[tid] = b3[tid];
    if (tid < 16) sB4[tid] = b4[tid];
    __syncthreads();

    const int wave  = tid >> 5;
    const int lane  = tid & 31;
    const int lrow  = lane & 15;
    const int khalf = lane >> 4;
    _Float16* myH3  = sH3[wave];

    const int tile = blockIdx.x * WPB + wave;   // 0..255, exact
    const int g0 = tile << 4;
    const float* grow = gacc + (size_t)(g0 + lrow) * GSD;

    v16h a1[4];
    #pragma unroll
    for (int kt = 0; kt < 4; ++kt) {
        const float* p = grow + kt * 32 + khalf * 8;
        v4f r0 = *(const v4f*)(p);
        v4f r1 = *(const v4f*)(p + 4);
        v4f r2 = *(const v4f*)(p + 16);
        v4f r3 = *(const v4f*)(p + 20);
        v16h a;
        #pragma unroll
        for (int j = 0; j < 4; ++j) {
            a[j]      = (_Float16)r0[j];
            a[4 + j]  = (_Float16)r1[j];
            a[8 + j]  = (_Float16)r2[j];
            a[12 + j] = (_Float16)r3[j];
        }
        a1[kt] = a;
    }

    #pragma unroll
    for (int nt = 0; nt < 2; ++nt) {
        v8f acc = {};
        #pragma unroll
        for (int kt = 0; kt < 4; ++kt) {
            v16h b = *(const v16h*)&sW3[((nt << 2) | kt) * 512 + lane * 16];
            acc = wmma_f16(a1[kt], b, acc);
        }
        const float bias = sB3[nt * 16 + lrow];
        #pragma unroll
        for (int v = 0; v < 8; ++v) {
            float h = fmaxf(acc[v] + bias, 0.0f);
            myH3[(v + 8 * khalf) * 32 + nt * 16 + lrow] = (_Float16)h;
        }
    }

    v16h a4;
    {
        const _Float16* p = myH3 + lrow * 32 + khalf * 8;
        v8h lo = *(const v8h*)(p);
        v8h hi = *(const v8h*)(p + 16);
        #pragma unroll
        for (int j = 0; j < 8; ++j) { a4[j] = lo[j]; a4[8 + j] = hi[j]; }
    }

    v8f acc = {};
    acc = wmma_f16(a4, *(const v16h*)&sW4[lane * 16], acc);
    const float bias = sB4[lrow];
    #pragma unroll
    for (int v = 0; v < 8; ++v)
        out[(size_t)(g0 + v + 8 * khalf) * 16 + lrow] = acc[v] + bias;
}

// ---------------------------------------------------------------------------
extern "C" void kernel_launch(void* const* d_in, const int* in_sizes, int n_in,
                              void* d_out, int out_size, void* d_ws, size_t ws_size,
                              hipStream_t stream) {
    const float* X  = (const float*)d_in[0];
    const int* gidx = (const int*)d_in[1];
    const float* W1 = (const float*)d_in[2];
    const float* b1 = (const float*)d_in[3];
    const float* W2 = (const float*)d_in[4];
    const float* b2 = (const float*)d_in[5];
    const float* W3 = (const float*)d_in[6];
    const float* b3 = (const float*)d_in[7];
    const float* W4 = (const float*)d_in[8];
    const float* b4 = (const float*)d_in[9];
    float* out  = (float*)d_out;
    float* gacc = (float*)d_ws;   // 4096*128 f32 = 2 MB accumulator

    const int nAcc = N_GRAPHS_C * GSD;
    zero_f32<<<(nAcc + 255) / 256, 256, 0, stream>>>(gacc, nAcc);

    mlp1_gate_scatter<<<1024, BLOCK_SZ, 0, stream>>>(X, gidx, W1, b1, W2, b2, gacc);

    mlp2_readout<<<(N_GRAPHS_C / 16) / WPB, BLOCK_SZ, 0, stream>>>(gacc, W3, b3, W4, b4, out);
}
